// GIN_73057393705211
// MI455X (gfx1250) — compile-verified, matrix operand order checked by reading, not run on hardware
//
#include <hip/hip_runtime.h>
#include <hip/hip_bf16.h>
#include <math.h>

#define N_NODES   50000
#define N_EDGES   800000
#define N_GRAPHS  512
#define IN_DIM    126
#define DIM_H     128
#define LIN1_DIM  64
#define BN_EPS    1e-5f
#define NTILES    (N_NODES / 16)   /* 3125, exact */

typedef __attribute__((ext_vector_type(16))) _Float16 v16h;
typedef __attribute__((ext_vector_type(8)))  _Float16 v8h;
typedef __attribute__((ext_vector_type(4)))  _Float16 v4h;
typedef __attribute__((ext_vector_type(8)))  float    v8f;

__device__ __forceinline__ void atomAddF32(float* p, float v) {
  __hip_atomic_fetch_add(p, v, __ATOMIC_RELAXED, __HIP_MEMORY_SCOPE_AGENT);
}

// ---------------------------------------------------------------------------
// Pack a row-major f32 weight matrix [din x 128] into the WMMA B-operand
// per-lane layout (f16). Layout per ISA 7.12.2 (16-bit operand, wave32):
//   lane l holds column N = l%16; half index v in the lane's v16h maps to
//   K(v,l) = kc*32 + (v&7) + (v>>3)*16 + (l>=16 ? 8 : 0).
// Packed order: [kc(4)][nt(8)][lane(32)][v(16)] halfs  -> 16384 halfs = 32KB.
// Rows K >= din are zero-padded (handles IN_DIM=126).
// ---------------------------------------------------------------------------
__global__ void pack_weights_kernel(const float* __restrict__ w, int din,
                                    _Float16* __restrict__ wp) {
  int idx = blockIdx.x * blockDim.x + threadIdx.x;
  if (idx >= 16384) return;
  int v    = idx & 15;
  int lane = (idx >> 4) & 31;
  int nt   = (idx >> 9) & 7;
  int kc   = idx >> 12;
  int K = kc * 32 + (v & 7) + ((v >> 3) << 4) + ((lane >= 16) ? 8 : 0);
  int J = nt * 16 + (lane & 15);
  wp[idx] = (K < din) ? (_Float16)w[K * DIM_H + J] : (_Float16)0.f;
}

// BN fused epilogue params: relu((X@W)*sc + sh)
__global__ void param_bn_kernel(const float* __restrict__ gamma,
                                const float* __restrict__ beta,
                                const float* __restrict__ mean,
                                const float* __restrict__ var,
                                const float* __restrict__ b1,
                                float* __restrict__ sc, float* __restrict__ sh) {
  int j = threadIdx.x;
  if (j >= DIM_H) return;
  float s = gamma[j] * rsqrtf(var[j] + BN_EPS);
  sc[j] = s;
  sh[j] = beta[j] + (b1[j] - mean[j]) * s;
}

__global__ void param_id_kernel(const float* __restrict__ b2,
                                float* __restrict__ sc, float* __restrict__ sh) {
  int j = threadIdx.x;
  if (j >= DIM_H) return;
  sc[j] = 1.f;
  sh[j] = b2[j];
}

// ---------------------------------------------------------------------------
// Edge scatter-add: agg[dst] += x[src].  One wave per edge, 4 cols per lane.
// ---------------------------------------------------------------------------
__global__ void scatter_f32_kernel(const float* __restrict__ x,  // [N, IN_DIM]
                                   const int* __restrict__ src,
                                   const int* __restrict__ dst,
                                   float* __restrict__ agg) {     // [N, 128]
  int e = blockIdx.x * 8 + (threadIdx.x >> 5);
  if (e >= N_EDGES) return;
  int lane = threadIdx.x & 31;
  int s = src[e], d = dst[e];
  int c0 = lane * 4;
  const float* xr = x + (long)s * IN_DIM;
  float* ar = agg + (long)d * DIM_H;
#pragma unroll
  for (int k = 0; k < 4; ++k) {
    int c = c0 + k;
    if (c < IN_DIM) atomAddF32(ar + c, xr[c]);
  }
}

__global__ void scatter_f16_kernel(const _Float16* __restrict__ h, // [N, 128]
                                   const int* __restrict__ src,
                                   const int* __restrict__ dst,
                                   float* __restrict__ agg) {      // [N, 128]
  int e = blockIdx.x * 8 + (threadIdx.x >> 5);
  if (e >= N_EDGES) return;
  int lane = threadIdx.x & 31;
  int s = src[e], d = dst[e];
  v4h hv = *(const v4h*)(h + (long)s * DIM_H + lane * 4);
  float* ar = agg + (long)d * DIM_H + lane * 4;
#pragma unroll
  for (int k = 0; k < 4; ++k) atomAddF32(ar + k, (float)hv[k]);
}

// y16 = f16(x + agg), zero-padded to 128 cols
__global__ void prep_f32_kernel(const float* __restrict__ x,      // [N, IN_DIM]
                                const float* __restrict__ agg,    // [N, 128]
                                _Float16* __restrict__ y16) {     // [N, 128]
  int idx = blockIdx.x * blockDim.x + threadIdx.x;
  if (idx >= N_NODES * DIM_H) return;
  int c = idx & (DIM_H - 1);
  int i = idx >> 7;
  float xv = (c < IN_DIM) ? x[(long)i * IN_DIM + c] : 0.f;
  y16[idx] = (_Float16)(xv + agg[idx]);
}

__global__ void prep_f16_kernel(const _Float16* __restrict__ h,   // [N, 128]
                                const float* __restrict__ agg,    // [N, 128]
                                _Float16* __restrict__ y16) {     // [N, 128]
  int idx = blockIdx.x * blockDim.x + threadIdx.x;
  if (idx >= N_NODES * DIM_H) return;
  y16[idx] = (_Float16)((float)h[idx] + agg[idx]);
}

// ---------------------------------------------------------------------------
// GEMM: O = relu((A @ W) * sc + sh), A:[N,128] f16, W packed 32KB, O:[N,128] f16
// 4 waves / block, each wave computes a 16x128 tile with 32 WMMA ops.
// ---------------------------------------------------------------------------
__global__ __launch_bounds__(128) void gemm_relu_kernel(
    const _Float16* __restrict__ A,
    const _Float16* __restrict__ Wp,
    const float* __restrict__ sc, const float* __restrict__ sh,
    _Float16* __restrict__ O) {
  __shared__ _Float16 lds_w[16384];   // 32 KB packed weights
  {
    const float4* s = (const float4*)Wp;
    float4* dst = (float4*)lds_w;
    for (int i = threadIdx.x; i < 2048; i += 128) dst[i] = s[i];
  }
  __syncthreads();

  int wave = threadIdx.x >> 5;
  int lane = threadIdx.x & 31;
  int tile = blockIdx.x * 4 + wave;
  bool active = tile < NTILES;
  int row0 = active ? tile * 16 : 0;

  int m    = lane & 15;           // A row / D column within tile
  int koff = (lane >> 4) << 3;    // 0 or 8 (K sub-offset per half-wave)
  int mo   = (lane >> 4) << 3;    // D row offset per half-wave

  const _Float16* arow = A + (long)(row0 + m) * DIM_H;
  v16h a[4];
#pragma unroll
  for (int kc = 0; kc < 4; ++kc) {
    v8h lo = *(const v8h*)(arow + kc * 32 + koff);
    v8h hi = *(const v8h*)(arow + kc * 32 + koff + 16);
    a[kc] = __builtin_shufflevector(lo, hi, 0, 1, 2, 3, 4, 5, 6, 7,
                                    8, 9, 10, 11, 12, 13, 14, 15);
  }

#pragma unroll
  for (int nt = 0; nt < 8; ++nt) {
    v8f acc = {};
#pragma unroll
    for (int kc = 0; kc < 4; ++kc) {
      v16h b = *(const v16h*)(lds_w + (((kc * 8 + nt) * 32 + lane) << 4));
      acc = __builtin_amdgcn_wmma_f32_16x16x32_f16(
          false, a[kc], false, b, (short)0, acc, false, false);
    }
    int j = nt * 16 + m;
    float s = sc[j], t = sh[j];
    if (active) {
#pragma unroll
      for (int r = 0; r < 8; ++r) {
        float v = acc[r] * s + t;
        v = v > 0.f ? v : 0.f;
        O[(long)(row0 + mo + r) * DIM_H + j] = (_Float16)v;
      }
    }
  }
}

// global_add_pool: hg[batch[i]] += h[i]
__global__ void pool_kernel(const _Float16* __restrict__ h,   // [N, 128]
                            const int* __restrict__ batch,
                            float* __restrict__ hg) {          // [G, 128]
  int idx = blockIdx.x * blockDim.x + threadIdx.x;
  if (idx >= N_NODES * DIM_H) return;
  int c = idx & (DIM_H - 1);
  int i = idx >> 7;
  int g = batch[i];
  atomAddF32(hg + (long)g * DIM_H + c, (float)h[idx]);
}

// final MLP: sigmoid(relu(hg@l1 + b1) @ l2 + b2), one block of 64 per graph
__global__ __launch_bounds__(64) void mlp_kernel(
    const float* __restrict__ hg,
    const float* __restrict__ l1w,  // [128, 64]
    const float* __restrict__ l1b,  // [64]
    const float* __restrict__ l2w,  // [64, 1]
    const float* __restrict__ l2b,  // [1]
    float* __restrict__ out) {      // [G]
  __shared__ float red[LIN1_DIM];
  int g = blockIdx.x;
  int t = threadIdx.x;
  const float* hr = hg + (long)g * DIM_H;
  float acc = l1b[t];
  for (int k = 0; k < DIM_H; ++k) acc += hr[k] * l1w[k * LIN1_DIM + t];
  acc = acc > 0.f ? acc : 0.f;
  red[t] = acc * l2w[t];
  __syncthreads();
  if (t == 0) {
    float z = l2b[0];
    for (int k = 0; k < LIN1_DIM; ++k) z += red[k];
    out[g] = 1.f / (1.f + __expf(-z));
  }
}

// ---------------------------------------------------------------------------
static inline void* bump(char*& p, size_t bytes) {
  void* r = (void*)p;
  p += (bytes + 255) & ~(size_t)255;
  return r;
}

extern "C" void kernel_launch(void* const* d_in, const int* in_sizes, int n_in,
                              void* d_out, int out_size, void* d_ws, size_t ws_size,
                              hipStream_t stream) {
  (void)in_sizes; (void)n_in; (void)out_size; (void)ws_size;
  const float* x     = (const float*)d_in[0];
  const int*   eidx  = (const int*)d_in[1];
  const int*   batch = (const int*)d_in[2];
  const int*   esrc  = eidx;
  const int*   edst  = eidx + N_EDGES;

  // per-conv param blocks: base 3, 11, 19 ; offsets w1,b1,gamma,beta,mean,var,w2,b2
  const int cbase[3] = {3, 11, 19};
  const float* l1w = (const float*)d_in[27];
  const float* l1b = (const float*)d_in[28];
  const float* l2w = (const float*)d_in[29];
  const float* l2b = (const float*)d_in[30];

  char* wsp = (char*)d_ws;
  float*    agg  = (float*)bump(wsp, (size_t)N_NODES * DIM_H * 4);
  _Float16* y16  = (_Float16*)bump(wsp, (size_t)N_NODES * DIM_H * 2);
  _Float16* t16  = (_Float16*)bump(wsp, (size_t)N_NODES * DIM_H * 2);
  _Float16* hA   = (_Float16*)bump(wsp, (size_t)N_NODES * DIM_H * 2);
  _Float16* hB   = (_Float16*)bump(wsp, (size_t)N_NODES * DIM_H * 2);
  _Float16* wp[6];
  for (int i = 0; i < 6; ++i) wp[i] = (_Float16*)bump(wsp, 16384 * 2);
  float* sc[6]; float* sh[6];
  for (int i = 0; i < 6; ++i) {
    sc[i] = (float*)bump(wsp, DIM_H * 4);
    sh[i] = (float*)bump(wsp, DIM_H * 4);
  }
  float* hg = (float*)bump(wsp, (size_t)N_GRAPHS * DIM_H * 4);

  // ---- pack all 6 weight matrices + epilogue params -----------------------
  for (int L = 0; L < 3; ++L) {
    int b = cbase[L];
    const float* w1    = (const float*)d_in[b + 0];
    const float* b1    = (const float*)d_in[b + 1];
    const float* gamma = (const float*)d_in[b + 2];
    const float* beta  = (const float*)d_in[b + 3];
    const float* mean  = (const float*)d_in[b + 4];
    const float* var   = (const float*)d_in[b + 5];
    const float* w2    = (const float*)d_in[b + 6];
    const float* b2    = (const float*)d_in[b + 7];
    int din1 = (L == 0) ? IN_DIM : DIM_H;
    pack_weights_kernel<<<64, 256, 0, stream>>>(w1, din1, wp[2 * L + 0]);
    pack_weights_kernel<<<64, 256, 0, stream>>>(w2, DIM_H, wp[2 * L + 1]);
    param_bn_kernel<<<1, 128, 0, stream>>>(gamma, beta, mean, var, b1,
                                           sc[2 * L + 0], sh[2 * L + 0]);
    param_id_kernel<<<1, 128, 0, stream>>>(b2, sc[2 * L + 1], sh[2 * L + 1]);
  }

  const int nelem   = N_NODES * DIM_H;
  const int eblocks = (N_EDGES + 7) / 8;
  const int nblocks = (nelem + 255) / 256;
  const int gblocks = (NTILES + 3) / 4;

  // ---- layer 1 (input: f32 x, 126 cols) -----------------------------------
  hipMemsetAsync(agg, 0, (size_t)nelem * 4, stream);
  scatter_f32_kernel<<<eblocks, 256, 0, stream>>>(x, esrc, edst, agg);
  prep_f32_kernel<<<nblocks, 256, 0, stream>>>(x, agg, y16);
  gemm_relu_kernel<<<gblocks, 128, 0, stream>>>(y16, wp[0], sc[0], sh[0], t16);
  gemm_relu_kernel<<<gblocks, 128, 0, stream>>>(t16, wp[1], sc[1], sh[1], hA);

  // ---- layer 2 ------------------------------------------------------------
  hipMemsetAsync(agg, 0, (size_t)nelem * 4, stream);
  scatter_f16_kernel<<<eblocks, 256, 0, stream>>>(hA, esrc, edst, agg);
  prep_f16_kernel<<<nblocks, 256, 0, stream>>>(hA, agg, y16);
  gemm_relu_kernel<<<gblocks, 128, 0, stream>>>(y16, wp[2], sc[2], sh[2], t16);
  gemm_relu_kernel<<<gblocks, 128, 0, stream>>>(t16, wp[3], sc[3], sh[3], hB);

  // ---- layer 3 ------------------------------------------------------------
  hipMemsetAsync(agg, 0, (size_t)nelem * 4, stream);
  scatter_f16_kernel<<<eblocks, 256, 0, stream>>>(hB, esrc, edst, agg);
  prep_f16_kernel<<<nblocks, 256, 0, stream>>>(hB, agg, y16);
  gemm_relu_kernel<<<gblocks, 128, 0, stream>>>(y16, wp[4], sc[4], sh[4], t16);
  gemm_relu_kernel<<<gblocks, 128, 0, stream>>>(t16, wp[5], sc[5], sh[5], hA);

  // ---- pooling + head -----------------------------------------------------
  hipMemsetAsync(hg, 0, (size_t)N_GRAPHS * DIM_H * 4, stream);
  pool_kernel<<<nblocks, 256, 0, stream>>>(hA, batch, hg);
  mlp_kernel<<<N_GRAPHS, 64, 0, stream>>>(hg, l1w, l1b, l2w, l2b, (float*)d_out);
}
